// _MultiHeadAttention_51256139710715
// MI455X (gfx1250) — matrix operand order repaired and checked by reading, not compiled, not run on hardware
//
#include <hip/hip_runtime.h>
#include <hip/hip_bf16.h>

#define BD 4
#define TD 1024
#define CD 256
#define HD 8
#define DD 32
#define WD 128
#define SPP 1152            // padded stride for (T + W - 1 = 1151)
#define KVS 264             // kvo/window row stride in bf16 (256 + 8 pad)

typedef __attribute__((ext_vector_type(16))) __bf16 v16bf;
typedef __attribute__((ext_vector_type(8)))  float  v8f;
typedef __attribute__((ext_vector_type(4)))  unsigned int u32x4;
typedef __attribute__((ext_vector_type(4)))  int i32x4v;
typedef __attribute__((ext_vector_type(8)))  int i32x8v;

union Frag {
  v16bf v;
  unsigned short u[16];
  uint4 q[2];
};

static __device__ __forceinline__ unsigned short f2bf(float f) {
  unsigned u = __float_as_uint(f);
  u += 0x7fffu + ((u >> 16) & 1u);          // round-to-nearest-even
  return (unsigned short)(u >> 16);
}

// ---------- TDM: 2D tile load global -> LDS (D# per CDNA5 ISA ch.8) ----------
// g1d0 packs: workgroup_mask=0 | data_size | pad_enable | pad_interval | pad_amount
static __device__ __forceinline__ void tdm_load_2d(
    const void* gsrc, unsigned lds_off, unsigned g1d0,
    unsigned ten_d0, unsigned ten_d1, unsigned til_d0, unsigned til_d1,
    unsigned stride0)
{
  unsigned long long ga = (unsigned long long)(size_t)gsrc;
  u32x4 g0;
  g0[0] = 1u;                                              // count=1, user mode
  g0[1] = lds_off;                                         // LDS byte address
  g0[2] = (unsigned)ga;                                    // global_addr[31:0]
  g0[3] = (unsigned)((ga >> 32) & 0x01ffffffu) | 0x80000000u; // addr[56:32] | type=2
  i32x8v g1;
  g1[0] = (int)g1d0;
  g1[1] = (int)((ten_d0 & 0xffffu) << 16);                 // tensor_dim0 lo
  g1[2] = (int)(((ten_d0 >> 16) & 0xffffu) | ((ten_d1 & 0xffffu) << 16));
  g1[3] = (int)(((ten_d1 >> 16) & 0xffffu) | ((til_d0 & 0xffffu) << 16));
  g1[4] = (int)(til_d1 & 0xffffu);                         // tile_dim1 (tile_dim2=0)
  g1[5] = (int)stride0;                                    // tensor_dim0_stride lo
  g1[6] = 0;
  g1[7] = 0;
  i32x4v z4 = {0, 0, 0, 0};
#if __has_include(<hip/amd_detail/amd_gfx1250_TDM.h>)
  i32x8v z8 = {0, 0, 0, 0, 0, 0, 0, 0};
  __builtin_amdgcn_tensor_load_to_lds(g0, g1, z4, z4, z8, 0);
#else
  __builtin_amdgcn_tensor_load_to_lds(g0, g1, z4, z4, 0);
#endif
}
static __device__ __forceinline__ unsigned lds_off_of(const void* p) {
  return (unsigned)(size_t)p;   // low 32 bits of flat shared pointer = LDS offset
}

// ---- prepass 1: xT[b][ch][sp] = bf16(xp[b][sp][ch]), left-padded by W-1 zeros
__global__ __launch_bounds__(256)
void pack_x_kernel(const float* __restrict__ x, unsigned short* __restrict__ xT) {
  int i = blockIdx.x * 256 + threadIdx.x;     // over B*C*SPP
  if (i >= BD * CD * SPP) return;
  int sp = i % SPP;
  int r  = i / SPP;
  int ch = r % CD;
  int b  = r / CD;
  int s  = sp - (WD - 1);
  float val = 0.f;
  if (sp < (TD + WD - 1) && s >= 0) val = x[(b * TD + s) * CD + ch];
  xT[i] = f2bf(val);
}

// ---- prepass 2: pack Wq/Wk/Wv/Wo transposes into WMMA B-fragment lane order
__global__ __launch_bounds__(256)
void pack_w_kernel(const float* __restrict__ Wq, const float* __restrict__ Wk,
                   const float* __restrict__ Wv, const float* __restrict__ Wo,
                   unsigned short* __restrict__ pkW) {
  int i = blockIdx.x * 256 + threadIdx.x;     // 4 * 65536
  if (i >= 4 * 65536) return;
  int wi = i >> 16;
  const float* W = (wi == 0) ? Wq : (wi == 1) ? Wk : (wi == 2) ? Wv : Wo;
  int p  = i & 65535;
  int e  = p & 15;
  int l  = (p >> 4) & 31;
  int ks = (p >> 9) & 7;
  int nt = p >> 12;
  int n = nt * 16 + (l & 15);
  int k = ks * 32 + e + 16 * (l >> 4);
  pkW[i] = f2bf(W[n * CD + k]);
}

// B fragment straight from global packed weights (coalesced)
static __device__ __forceinline__ void ldgB(Frag& f, const unsigned short* pk,
                                            int nt, int ks, int lane) {
  const unsigned short* bp = pk + ((nt * 8 + ks) * 32 + lane) * 16;
  f.q[0] = *(const uint4*)bp;
  f.q[1] = *(const uint4*)(bp + 8);
}
// B fragment from a TDM-staged 2-K-step LDS panel (nt-major, 1024 elems per nt)
static __device__ __forceinline__ void ldsB2(Frag& f, const unsigned short* buf,
                                             int ksl, int nt, int lane) {
  const unsigned short* bp = buf + nt * 1024 + ksl * 512 + lane * 16;
  f.q[0] = *(const uint4*)bp;
  f.q[1] = *(const uint4*)(bp + 8);
}
// A fragment from an LDS row; per-lane k is a contiguous 16-element run (k = 16*half + s)
static __device__ __forceinline__ void ldsA(Frag& f, const unsigned short* row,
                                            int ks, int half) {
  const unsigned short* p = row + ks * 32 + 16 * half;
  f.q[0] = *(const uint4*)p;
  f.q[1] = *(const uint4*)(p + 8);
}

// ---- main: one block per (b,t)
__global__ __launch_bounds__(256)
void attn_kernel(const float* __restrict__ bq, const float* __restrict__ bk,
                 const float* __restrict__ bv, const float* __restrict__ bo,
                 const unsigned short* __restrict__ xT,
                 const unsigned short* __restrict__ pkW,
                 float* __restrict__ out)
{
  extern __shared__ unsigned char smem[];
  unsigned short* Bs  = (unsigned short*)smem;   // 2 x 16384 bf16 ping-pong panels
  unsigned short* kvo = Bs + 32768;              // 128 x 264 bf16: window, K, then V
  float* qv   = (float*)(kvo + WD * KVS);        // 256 f32
  float* attn = qv + CD;                         // 8 x 128 f32
  unsigned short* xrb = (unsigned short*)(attn + HD * WD); // 256 bf16 token row
  unsigned short* ocb = xrb + CD;                // 256 bf16 attn output

  const int t    = blockIdx.x;
  const int b    = blockIdx.y;
  const int tid  = threadIdx.x;
  const int lane = tid & 31;
  const int mt   = tid >> 5;      // wave id
  const int half = lane >> 4;
  const int ln16 = lane & 15;

  const unsigned short* xTb = xT + b * CD * SPP;
  const unsigned bs_off  = lds_off_of(Bs);
  const unsigned kvo_off = lds_off_of(kvo);

  // ---- phase 0: kick off TDM DMAs (wave 0 only; TDM ignores EXEC)
  // window: 256 rows (channels) x 128 elems, row stride SPP; LDS pad 16B per 512B
  // -> lands exactly in kvo's 264-element padded row layout:
  //    kv[m][c] = xT[b][2m + (c>=128)][t + c%128]
  if (mt == 0) {
    tdm_load_2d(xTb + t, kvo_off,
                /*flags: size=2B, pad_en, interval=128dw(code6), amount=4dw(code3)*/
                0x00010000u | (1u << 20) | (6u << 22) | (3u << 25),
                SPP, 256, 128, 256, SPP);
    // first weight panel: Wk K-steps 0..1 (16 nt-rows x 1024 elems, stride 4096)
    tdm_load_2d(pkW + (size_t)1 * 65536, bs_off,
                0x00010000u, 4096, 16, 1024, 16, 4096);
  }
  if (mt == 0) __builtin_amdgcn_s_wait_tensorcnt(1);   // window done (in-order)
  __syncthreads();

  // token row x[b,t,ch] lives at window columns 127 / 255 of row ch>>1
  xrb[tid] = kvo[(tid >> 1) * KVS + 127 + 128 * (tid & 1)];
  __syncthreads();

  // ---- phase 2: hoist K/V A-fragments (before kvo is overwritten),
  //               q projection via WMMA (A rows replicated with the token row)
  Frag kvA[8];
  {
    const unsigned short* row = kvo + (mt * 16 + ln16) * KVS;
    #pragma unroll
    for (int ks = 0; ks < 8; ++ks) ldsA(kvA[ks], row, ks, half);
  }
  {
    Frag aq[8];
    #pragma unroll
    for (int ks = 0; ks < 8; ++ks) ldsA(aq[ks], xrb, ks, half);
    v8f qacc[2];
    #pragma unroll
    for (int i = 0; i < 2; ++i)
      #pragma unroll
      for (int r = 0; r < 8; ++r) qacc[i][r] = 0.f;
    #pragma unroll
    for (int i = 0; i < 2; ++i) {
      const int nt = mt * 2 + i;
      #pragma unroll
      for (int ks = 0; ks < 8; ++ks) {
        Frag bf;
        ldgB(bf, pkW, nt, ks, lane);   // pkW[0] = Wq
        qacc[i] = __builtin_amdgcn_wmma_f32_16x16x32_bf16(
            false, aq[ks].v, false, bf.v, (short)0, qacc[i], false, false);
      }
    }
    if (lane < 16) {
      #pragma unroll
      for (int i = 0; i < 2; ++i) {
        int n = (mt * 2 + i) * 16 + lane;
        qv[n] = qacc[i][0] + bq[n];    // D row 0 = q (all A rows identical)
      }
    }
  }

  // ---- phase 3: K then V projection GEMMs (128x256x256), TDM double-buffered
  for (int proj = 0; proj < 2; ++proj) {
    const unsigned short* pk = pkW + (size_t)(proj + 1) * 65536;  // Wk, Wv
    const float* bias = proj ? bv : bk;

    v8f acc[16];
    #pragma unroll
    for (int nt = 0; nt < 16; ++nt)
      #pragma unroll
      for (int r = 0; r < 8; ++r) acc[nt][r] = 0.f;

    #pragma unroll
    for (int s = 0; s < 4; ++s) {            // 4 stages of 2 K-steps
      if (mt == 0) __builtin_amdgcn_s_wait_tensorcnt(0);
      __syncthreads();
      if (mt == 0) {
        if (s < 3)          // prefetch next panel into the other buffer
          tdm_load_2d(pk + 1024 * (s + 1), bs_off + 32768u * ((s + 1) & 1),
                      0x00010000u, 4096, 16, 1024, 16, 4096);
        else if (proj == 0) // prefetch V panel 0 during K-store/scores/softmax
          tdm_load_2d(pkW + (size_t)2 * 65536, bs_off,
                      0x00010000u, 4096, 16, 1024, 16, 4096);
      }
      const unsigned short* Bb = Bs + 16384 * (s & 1);
      #pragma unroll
      for (int ksl = 0; ksl < 2; ++ksl) {
        const int ks = s * 2 + ksl;
        Frag b0, b1;
        ldsB2(b0, Bb, ksl, 0, lane);
        #pragma unroll
        for (int nt = 0; nt < 16; nt += 2) {
          ldsB2(b1, Bb, ksl, nt + 1, lane);
          acc[nt] = __builtin_amdgcn_wmma_f32_16x16x32_bf16(
              false, kvA[ks].v, false, b0.v, (short)0, acc[nt], false, false);
          if (nt + 2 < 16) ldsB2(b0, Bb, ksl, nt + 2, lane);
          acc[nt + 1] = __builtin_amdgcn_wmma_f32_16x16x32_bf16(
              false, kvA[ks].v, false, b1.v, (short)0, acc[nt + 1], false, false);
        }
      }
    }

    __syncthreads();   // previous consumers of kvo done before overwrite
    #pragma unroll
    for (int nt = 0; nt < 16; ++nt) {
      float bb = bias[nt * 16 + ln16];
      #pragma unroll
      for (int r = 0; r < 8; ++r) {
        int m = mt * 16 + r + 8 * half;
        kvo[m * KVS + nt * 16 + ln16] = f2bf(acc[nt][r] + bb);
      }
    }
    __syncthreads();

    if (proj == 0) {
      // scores[h][w] = (q_h . k_w_h)/sqrt(32), masked where w + t < 127
      int h  = tid >> 5;
      int wb = (tid & 31) * 4;
      #pragma unroll
      for (int w = wb; w < wb + 4; ++w) {
        const uint4* kp = (const uint4*)(kvo + w * KVS + h * DD);
        float s = 0.f;
        #pragma unroll
        for (int j = 0; j < 4; ++j) {
          uint4 u = kp[j];
          const float* qp = qv + h * DD + j * 8;
          s += qp[0] * __uint_as_float(u.x << 16);
          s += qp[1] * __uint_as_float(u.x & 0xffff0000u);
          s += qp[2] * __uint_as_float(u.y << 16);
          s += qp[3] * __uint_as_float(u.y & 0xffff0000u);
          s += qp[4] * __uint_as_float(u.z << 16);
          s += qp[5] * __uint_as_float(u.z & 0xffff0000u);
          s += qp[6] * __uint_as_float(u.w << 16);
          s += qp[7] * __uint_as_float(u.w & 0xffff0000u);
        }
        s *= 0.17677669529663687f;   // 1/sqrt(32)
        if (w + t < WD - 1) s = -__builtin_inff();
        attn[h * WD + w] = s;
      }
      __syncthreads();
      { // softmax: wave mt owns head mt; lane owns 4 consecutive w
        float v0 = attn[mt * WD + lane * 4 + 0];
        float v1 = attn[mt * WD + lane * 4 + 1];
        float v2 = attn[mt * WD + lane * 4 + 2];
        float v3 = attn[mt * WD + lane * 4 + 3];
        float mx = fmaxf(fmaxf(v0, v1), fmaxf(v2, v3));
        #pragma unroll
        for (int off = 16; off > 0; off >>= 1) mx = fmaxf(mx, __shfl_xor(mx, off, 32));
        v0 = __expf(v0 - mx); v1 = __expf(v1 - mx);
        v2 = __expf(v2 - mx); v3 = __expf(v3 - mx);
        float sum = v0 + v1 + v2 + v3;
        #pragma unroll
        for (int off = 16; off > 0; off >>= 1) sum += __shfl_xor(sum, off, 32);
        float inv = 1.f / sum;
        attn[mt * WD + lane * 4 + 0] = v0 * inv;
        attn[mt * WD + lane * 4 + 1] = v1 * inv;
        attn[mt * WD + lane * 4 + 2] = v2 * inv;
        attn[mt * WD + lane * 4 + 3] = v3 * inv;
      }
    }
  }

  // ---- phase 4: out[h,d] = sum_w attn[h,w] * v[w][h*32+d]  (lane-coalesced LDS)
  {
    int h = tid >> 5, d = tid & 31;
    float o = 0.f;
    #pragma unroll 8
    for (int w = 0; w < WD; ++w)
      o += attn[h * WD + w] *
           __uint_as_float((unsigned)kvo[w * KVS + h * DD + d] << 16);
    ocb[tid] = f2bf(o);
  }
  __syncthreads();

  // ---- phase 5: output projection via WMMA (A rows replicated with ocb)
  {
    Frag ao[8];
    #pragma unroll
    for (int ks = 0; ks < 8; ++ks) ldsA(ao[ks], ocb, ks, half);
    const unsigned short* pko = pkW + (size_t)3 * 65536;  // Wo
    v8f oacc[2];
    #pragma unroll
    for (int i = 0; i < 2; ++i)
      #pragma unroll
      for (int r = 0; r < 8; ++r) oacc[i][r] = 0.f;
    #pragma unroll
    for (int i = 0; i < 2; ++i) {
      const int nt = mt * 2 + i;
      #pragma unroll
      for (int ks = 0; ks < 8; ++ks) {
        Frag bf;
        ldgB(bf, pko, nt, ks, lane);
        oacc[i] = __builtin_amdgcn_wmma_f32_16x16x32_bf16(
            false, ao[ks].v, false, bf.v, (short)0, oacc[i], false, false);
      }
    }
    if (lane < 16) {
      #pragma unroll
      for (int i = 0; i < 2; ++i) {
        int n = (mt * 2 + i) * 16 + lane;
        out[(b * TD + t) * CD + n] = oacc[i][0] + bo[n];
      }
    }
  }
}

extern "C" void kernel_launch(void* const* d_in, const int* in_sizes, int n_in,
                              void* d_out, int out_size, void* d_ws, size_t ws_size,
                              hipStream_t stream) {
  (void)in_sizes; (void)n_in; (void)out_size; (void)ws_size;
  const float* x  = (const float*)d_in[0];
  const float* Wq = (const float*)d_in[1];
  const float* Wk = (const float*)d_in[2];
  const float* Wv = (const float*)d_in[3];
  const float* bq = (const float*)d_in[4];
  const float* bk = (const float*)d_in[5];
  const float* bv = (const float*)d_in[6];
  const float* Wo = (const float*)d_in[7];
  const float* bo = (const float*)d_in[8];
  float* out = (float*)d_out;

  unsigned short* xT  = (unsigned short*)d_ws;          // B*C*SPP bf16 = 2.25 MB
  unsigned short* pkW = xT + (size_t)BD * CD * SPP;     // 4*65536 bf16 = 512 KB

  int nx = BD * CD * SPP;
  pack_x_kernel<<<(nx + 255) / 256, 256, 0, stream>>>(x, xT);
  pack_w_kernel<<<(4 * 65536 + 255) / 256, 256, 0, stream>>>(Wq, Wk, Wv, Wo, pkW);

  // shared: (32768 + 128*264 + 256 + 256) bf16  +  (256 + 1024) f32 = 139,264 B
  size_t shmem = (size_t)(32768 + WD * KVS) * 2 +
                 (size_t)(CD + HD * WD) * 4 + (size_t)(CD + CD) * 2;
  dim3 grid(TD, BD);
  attn_kernel<<<grid, 256, shmem, stream>>>(bq, bk, bv, bo, xT, pkW, out);
}
